// VideoTokenizer_47416438948318
// MI455X (gfx1250) — matrix operand order repaired and checked by reading, not compiled, not run on hardware
//
#include <hip/hip_runtime.h>
#include <stdint.h>

// ---------------- problem constants (match reference) ----------------
#define B_    32
#define C_    2
#define T_    120
#define H_    72
#define W_    64
#define PT_   6
#define S_    72            // spatial patches per frame (9*8)
#define TOK_  1440          // tokens per batch element (20*72)
#define FLAT_ 768
#define D_    128
#define MT_   32            // patches per block (M tile)
#define NTHREADS 256
#define RSTR  776           // padded bf16 row stride for LDS A tile
#define KITERS 24           // 768 / 32
#define LN_EPS 1e-5f

typedef __attribute__((ext_vector_type(16))) __bf16 v16bf;
typedef __attribute__((ext_vector_type(8)))  float  v8f;

union Frag { v16bf v; uint4 q[2]; };

__device__ __forceinline__ unsigned short to_bf16(float a) {
  union { float f; unsigned u; } ua; ua.f = a;
  return (unsigned short)((ua.u + 0x7fffu + ((ua.u >> 16) & 1u)) >> 16);
}

__device__ __forceinline__ unsigned pack_bf16(float a, float b) {
  return (unsigned)to_bf16(a) | ((unsigned)to_bf16(b) << 16);
}

// ---------------------------------------------------------------------
// Pre-pass: fc_w (f32, [128][768] row-major) -> bf16 WMMA B-fragment
// layout in workspace:
//   wf[((k*8 + ntile)*32 + lane)*16 + j]
//   lane<16 : N = ntile*16+lane,    K = k*32 + (j&7) + 16*(j>>3)
//   lane>=16: N = ntile*16+lane-16, K = k*32 + 8 + (j&7) + 16*(j>>3)
// ---------------------------------------------------------------------
__global__ __launch_bounds__(256)
void vt_pack_w(const float* __restrict__ fcw, unsigned short* __restrict__ wf) {
  int i = blockIdx.x * 256 + threadIdx.x;      // 24*8*32*16 = 98304 total
  if (i >= 24 * 8 * 32 * 16) return;
  int j    =  i        & 15;
  int lane = (i >> 4)  & 31;
  int nt   = (i >> 9)  & 7;
  int k    =  i >> 12;                          // 0..23
  int d  = nt * 16 + (lane & 15);
  int kk = (j & 7) + ((lane < 16) ? 0 : 8) + ((j >> 3) << 4);
  int f  = k * 32 + kk;
  wf[i] = to_bf16(fcw[d * FLAT_ + f]);
}

// ---------------------------------------------------------------------
// Main fused kernel: patchify + LayerNorm + (bf16 WMMA) Linear 768->128
// ---------------------------------------------------------------------
__global__ __launch_bounds__(NTHREADS)
void vt_main(const float* __restrict__ x,
             const float* __restrict__ lnw,
             const float* __restrict__ lnb,
             const unsigned short* __restrict__ wf,
             const float* __restrict__ fcb,
             float* __restrict__ out) {
  __shared__ unsigned short A16[MT_ * RSTR];   // 49664 B, normalized bf16 A tile
  __shared__ float lnP[2 * FLAT_];             // 6144 B  (w then b)
  __shared__ float redS[4 * NTHREADS];         // 4096 B
  __shared__ float redQ[4 * NTHREADS];         // 4096 B
  __shared__ float stMu[MT_], stRs[MT_];       // 256 B   -> total 64256 B

  const int tid = threadIdx.x;
  const int p0  = blockIdx.x * MT_;            // first global patch of tile

  // stage LayerNorm params into LDS
  for (int i = tid; i < 384; i += NTHREADS) {
    float4 v = (i < 192) ? ((const float4*)lnw)[i] : ((const float4*)lnb)[i - 192];
    ((float4*)lnP)[i] = v;
  }

  // ---------------- pass 1: fp32 stats (sum, sumsq) ----------------
  float s0 = 0.f, s1 = 0.f, s2 = 0.f, s3 = 0.f;
  float q0 = 0.f, q1 = 0.f, q2 = 0.f, q3 = 0.f;
  for (int it = 0; it < 24; ++it) {
    int u    = tid + it * NTHREADS;            // 0..6143
    int dx4  =  u       & 1;
    int lmlo = (u >> 1) & 7;
    int dy   = (u >> 4) & 7;
    int v    =  u >> 7;
    int dt   = v % 6;
    int w2   = v / 6;
    int c    = w2 & 1;
    int lmhi = w2 >> 1;                        // wave-uniform
    int lm   = lmhi * 8 + lmlo;
    int p    = p0 + lm;
    int b    = p / TOK_;
    int qq   = p % TOK_;
    int t0   = qq / S_;
    int sp   = qq % S_;
    int ph   = sp >> 3, pw = sp & 7;
    int gi = (((b * C_ + c) * T_ + (t0 * PT_ + dt)) * H_ + (ph * 8 + dy)) * W_
             + pw * 8 + dx4 * 4;
    float4 xv = *(const float4*)(x + gi);
    float ss = xv.x + xv.y + xv.z + xv.w;
    float qs = fmaf(xv.x, xv.x, fmaf(xv.y, xv.y, fmaf(xv.z, xv.z, xv.w * xv.w)));
    if      (lmhi == 0) { s0 += ss; q0 += qs; }
    else if (lmhi == 1) { s1 += ss; q1 += qs; }
    else if (lmhi == 2) { s2 += ss; q2 += qs; }
    else                { s3 += ss; q3 += qs; }
  }
  redS[0 * NTHREADS + tid] = s0; redQ[0 * NTHREADS + tid] = q0;
  redS[1 * NTHREADS + tid] = s1; redQ[1 * NTHREADS + tid] = q1;
  redS[2 * NTHREADS + tid] = s2; redQ[2 * NTHREADS + tid] = q2;
  redS[3 * NTHREADS + tid] = s3; redQ[3 * NTHREADS + tid] = q3;
  __syncthreads();

  if (tid < MT_) {
    int lm = tid, lmhi = lm >> 3, lmlo = lm & 7;
    float s = 0.f, q = 0.f;
    #pragma unroll
    for (int m = 0; m < 16; ++m) {
      int t = lmlo * 2 + 16 * m;
      s += redS[lmhi * NTHREADS + t] + redS[lmhi * NTHREADS + t + 1];
      q += redQ[lmhi * NTHREADS + t] + redQ[lmhi * NTHREADS + t + 1];
    }
    float mu  = s * (1.0f / 768.0f);
    float var = q * (1.0f / 768.0f) - mu * mu;
    stMu[lm] = mu;
    stRs[lm] = rsqrtf(var + LN_EPS);
  }
  __syncthreads();

  // ------- pass 2: re-read (L2-hot), normalize, bf16 -> LDS A tile -------
  for (int it = 0; it < 24; ++it) {
    int u    = tid + it * NTHREADS;
    int dx4  =  u       & 1;
    int lmlo = (u >> 1) & 7;
    int dy   = (u >> 4) & 7;
    int v    =  u >> 7;
    int dt   = v % 6;
    int w2   = v / 6;
    int c    = w2 & 1;
    int lmhi = w2 >> 1;
    int lm   = lmhi * 8 + lmlo;
    int p    = p0 + lm;
    int b    = p / TOK_;
    int qq   = p % TOK_;
    int t0   = qq / S_;
    int sp   = qq % S_;
    int ph   = sp >> 3, pw = sp & 7;
    int gi = (((b * C_ + c) * T_ + (t0 * PT_ + dt)) * H_ + (ph * 8 + dy)) * W_
             + pw * 8 + dx4 * 4;
    float4 xv = *(const float4*)(x + gi);
    int f = c * 384 + dt * 64 + dy * 8 + dx4 * 4;
    float4 w4 = *(const float4*)&lnP[f];
    float4 b4 = *(const float4*)&lnP[FLAT_ + f];
    float mu = stMu[lm], rs = stRs[lm];
    float y0 = (xv.x - mu) * rs * w4.x + b4.x;
    float y1 = (xv.y - mu) * rs * w4.y + b4.y;
    float y2 = (xv.z - mu) * rs * w4.z + b4.z;
    float y3 = (xv.w - mu) * rs * w4.w + b4.w;
    uint2 pk;
    pk.x = pack_bf16(y0, y1);
    pk.y = pack_bf16(y2, y3);
    *(uint2*)&A16[lm * RSTR + f] = pk;
  }
  __syncthreads();

  // ---------------- GEMM via v_wmma_f32_16x16x32_bf16 ----------------
  const int lane = tid & 31;
  const int wv   = tid >> 5;                   // 0..7
  const int prow = (wv & 4) ? 16 : 0;          // which 16-patch half
  const int n0   = (wv & 3) * 2;               // two N tiles of 16
  const int n1   = n0 + 1;
  const int l15  = lane & 15;
  const int hsel = lane >> 4;                  // K-half selector per ISA A layout
  const unsigned short* arow = A16 + (prow + l15) * RSTR + hsel * 8;
  const uint4* wf4 = (const uint4*)wf;

  v8f acc0 = {};
  v8f acc1 = {};
  #pragma unroll 4
  for (int k = 0; k < KITERS; ++k) {
    Frag a, bf0, bf1;
    const uint4* ap = (const uint4*)(arow + k * 32);   // +64B per k-iter
    a.q[0] = ap[0];                                    // K 0-7   (or 8-15)
    a.q[1] = ap[2];                                    // K 16-23 (or 24-31)
    const uint4* bp0 = wf4 + (((k * 8 + n0) * 32 + lane) << 1);
    bf0.q[0] = bp0[0]; bf0.q[1] = bp0[1];
    const uint4* bp1 = wf4 + (((k * 8 + n1) * 32 + lane) << 1);
    bf1.q[0] = bp1[0]; bf1.q[1] = bp1[1];
    acc0 = __builtin_amdgcn_wmma_f32_16x16x32_bf16(false, a.v, false, bf0.v,
                                                   (short)0, acc0, false, false);
    acc1 = __builtin_amdgcn_wmma_f32_16x16x32_bf16(false, a.v, false, bf1.v,
                                                   (short)0, acc1, false, false);
  }

  // bias + store (C/D layout: lane<16 -> M=r, lane>=16 -> M=8+r; N=lane&15)
  float bias0 = fcb[n0 * 16 + l15];
  float bias1 = fcb[n1 * 16 + l15];
  int pb = p0 + prow + hsel * 8;
  #pragma unroll
  for (int r = 0; r < 8; ++r) {
    out[(pb + r) * D_ + n0 * 16 + l15] = acc0[r] + bias0;
    out[(pb + r) * D_ + n1 * 16 + l15] = acc1[r] + bias1;
  }
}

// ---------------------------------------------------------------------
extern "C" void kernel_launch(void* const* d_in, const int* in_sizes, int n_in,
                              void* d_out, int out_size, void* d_ws, size_t ws_size,
                              hipStream_t stream) {
  const float* x   = (const float*)d_in[0];
  const float* lnw = (const float*)d_in[1];
  const float* lnb = (const float*)d_in[2];
  const float* fcw = (const float*)d_in[3];
  const float* fcb = (const float*)d_in[4];
  float* out = (float*)d_out;
  unsigned short* wf = (unsigned short*)d_ws;   // 192 KB used

  vt_pack_w<<<384, 256, 0, stream>>>(fcw, wf);
  vt_main<<<(B_ * TOK_) / MT_, NTHREADS, 0, stream>>>(x, lnw, lnb, wf, fcb, out);
}
